// PointMAEPretrain_13615046328379
// MI455X (gfx1250) — matrix operand order repaired and checked by reading, not compiled, not run on hardware
//
#include <hip/hip_runtime.h>
#include <hip/hip_bf16.h>
#include <math.h>

typedef __bf16 bf16;
typedef __attribute__((ext_vector_type(16))) __bf16 v16bf;
typedef __attribute__((ext_vector_type(8)))  float  v8f;

#define B_    32
#define NPTS  2048
#define G_    128
#define KNN_  32
#define D_    384
#define GV    52
#define MT    (B_ * GV)        // 1664
#define DEPTH_ 12
#define CB    8                // batches per encoder chunk
#define CR    (CB * G_ * KNN_) // 32768 rows per chunk
#define CGRP  (CB * G_)        // 1024 groups per chunk
#define EPS_  1e-5f

// ---------------------------------------------------------------------------
// f32 -> bf16 bulk conversion (all weight tensors; element counts are
// multiples of 1024 so grids are exact).
// ---------------------------------------------------------------------------
__global__ __launch_bounds__(256) void cvt_bf16_k(
    const float* __restrict__ src, bf16* __restrict__ dst) {
  const int e = (blockIdx.x * 256 + threadIdx.x) * 4;
  float4 f = *(const float4*)(src + e);
  dst[e + 0] = (bf16)f.x; dst[e + 1] = (bf16)f.y;
  dst[e + 2] = (bf16)f.z; dst[e + 3] = (bf16)f.w;
}

// ---------------------------------------------------------------------------
// Farthest point sampling: one block per batch. Records `far` at iteration
// start (matches jax.lax.scan carry); argmax tie-break = first index.
// ---------------------------------------------------------------------------
__global__ __launch_bounds__(256) void fps_kernel(
    const float* __restrict__ xyz, int* __restrict__ cidx,
    float* __restrict__ center) {
  const int b = blockIdx.x, t = threadIdx.x;
  const float* px = xyz + (size_t)b * NPTS * 3;
  float lx[8], ly[8], lz[8], ld[8];
#pragma unroll
  for (int i = 0; i < 8; ++i) {
    int p = i * 256 + t;
    lx[i] = px[p * 3 + 0]; ly[i] = px[p * 3 + 1]; lz[i] = px[p * 3 + 2];
    ld[i] = 1e10f;
  }
  __shared__ float sv[256];
  __shared__ int   si[256];
  __shared__ float c3[3];
  __shared__ int   curFar;
  if (t == 0) curFar = 0;
  __syncthreads();
  for (int g = 0; g < G_; ++g) {
    int far = curFar;
    if (t == 0) cidx[b * G_ + g] = far;
    if (t < 3) {
      float c = px[far * 3 + t];
      c3[t] = c;
      center[((size_t)b * G_ + g) * 3 + t] = c;
    }
    __syncthreads();
    float cx = c3[0], cy = c3[1], cz = c3[2];
    float bv = -1.f; int bi = 0;
#pragma unroll
    for (int i = 0; i < 8; ++i) {
      float dx = lx[i] - cx, dy = ly[i] - cy, dz = lz[i] - cz;
      float d = dx * dx + dy * dy + dz * dz;
      ld[i] = fminf(ld[i], d);
      if (ld[i] > bv) { bv = ld[i]; bi = i * 256 + t; }
    }
    sv[t] = bv; si[t] = bi;
    __syncthreads();
    for (int o = 128; o > 0; o >>= 1) {
      if (t < o) {
        if (sv[t + o] > sv[t] || (sv[t + o] == sv[t] && si[t + o] < si[t])) {
          sv[t] = sv[t + o]; si[t] = si[t + o];
        }
      }
      __syncthreads();
    }
    if (t == 0) curFar = si[0];
    __syncthreads();
  }
}

// ---------------------------------------------------------------------------
// KNN (iterative min selection == stable top_k) fused with neighborhood
// gather + recenter. One block per (b, g).
// ---------------------------------------------------------------------------
__global__ __launch_bounds__(128) void knn_kernel(
    const float* __restrict__ xyz, const float* __restrict__ center,
    float* __restrict__ neigh) {
  const int bid = blockIdx.x;            // b*G_ + g
  const int b = bid / G_, t = threadIdx.x;
  const float* px = xyz + (size_t)b * NPTS * 3;
  const float cx = center[bid * 3 + 0], cy = center[bid * 3 + 1],
              cz = center[bid * 3 + 2];
  float dist[16];
#pragma unroll
  for (int i = 0; i < 16; ++i) {
    int p = i * 128 + t;
    float dx = px[p * 3 + 0] - cx, dy = px[p * 3 + 1] - cy,
          dz = px[p * 3 + 2] - cz;
    dist[i] = dx * dx + dy * dy + dz * dz;
  }
  __shared__ float sv[128];
  __shared__ int   si[128];
  __shared__ int   sel;
  for (int kk = 0; kk < KNN_; ++kk) {
    float bv = 3.4e38f; int bi = 0x7fffffff;
#pragma unroll
    for (int i = 0; i < 16; ++i)
      if (dist[i] < bv) { bv = dist[i]; bi = i * 128 + t; }
    sv[t] = bv; si[t] = bi;
    __syncthreads();
    for (int o = 64; o > 0; o >>= 1) {
      if (t < o) {
        if (sv[t + o] < sv[t] || (sv[t + o] == sv[t] && si[t + o] < si[t])) {
          sv[t] = sv[t + o]; si[t] = si[t + o];
        }
      }
      __syncthreads();
    }
    if (t == 0) sel = si[0];
    __syncthreads();
    const int j = sel;
    if ((j & 127) == t) dist[j >> 7] = 3.4e38f;
    if (t < 3)
      neigh[((size_t)bid * KNN_ + kk) * 3 + t] = px[j * 3 + t] - center[bid * 3 + t];
    __syncthreads();
  }
}

// ---------------------------------------------------------------------------
// Encoder stage 1: [rows,3] @ w1^T + b1, BN+ReLU -> bf16 [rows,128]
// ---------------------------------------------------------------------------
__global__ __launch_bounds__(256) void enc_stage1(
    const float* __restrict__ neigh, const float* __restrict__ w1,
    const float* __restrict__ b1, const float* __restrict__ g1,
    const float* __restrict__ bb1, const float* __restrict__ m1,
    const float* __restrict__ v1, bf16* __restrict__ X1, int rowBase) {
  const int e = blockIdx.x * 256 + threadIdx.x;   // < CR*128
  const int rl = e >> 7, c = e & 127;
  const float* nr = neigh + ((size_t)rowBase + rl) * 3;
  float x = nr[0] * w1[c * 3 + 0] + nr[1] * w1[c * 3 + 1] +
            nr[2] * w1[c * 3 + 2] + b1[c];
  x = (x - m1[c]) * rsqrtf(v1[c] + EPS_) * g1[c] + bb1[c];
  x = fmaxf(x, 0.f);
  X1[(size_t)rl * 128 + c] = (bf16)x;
}

// ---------------------------------------------------------------------------
// Max over the 32 points of each group. OUT_BF16: bf16 out (FG, lossless
// since inputs are bf16) else f32 (tokens).
// ---------------------------------------------------------------------------
template <bool OUT_BF16>
__global__ __launch_bounds__(256) void maxpool_k(
    const bf16* __restrict__ X, void* __restrict__ out, int C, int outOfs) {
  const int e = blockIdx.x * 256 + threadIdx.x;   // grp_local*C + c
  const int grp = e / C, c = e % C;
  const bf16* p = X + (size_t)grp * KNN_ * C + c;
  float m = -3.4e38f;
#pragma unroll 8
  for (int k = 0; k < KNN_; ++k) m = fmaxf(m, (float)p[(size_t)k * C]);
  const size_t o = ((size_t)(outOfs + grp)) * C + c;
  if (OUT_BF16) ((bf16*)out)[o] = (bf16)m;
  else          ((float*)out)[o] = m;
}

// ---------------------------------------------------------------------------
// Gather visible tokens -> h, gather visible centers.
// ---------------------------------------------------------------------------
__global__ __launch_bounds__(256) void gather_tokens(
    const float* __restrict__ tokens, const float* __restrict__ center,
    const int* __restrict__ vis, float* __restrict__ h,
    float* __restrict__ viscen) {
  const int e = blockIdx.x * 256 + threadIdx.x;   // < MT*D_
  const int r = e / D_, c = e % D_;
  const int b = r / GV;
  const size_t tok = (size_t)b * G_ + vis[r];
  h[e] = tokens[tok * D_ + c];
  if (c < 3) viscen[r * 3 + c] = center[tok * 3 + c];
}

// ---------------------------------------------------------------------------
// pos MLP first half: gelu([MT,3] @ pos_w1^T + b1) -> bf16 [MT,128]
// ---------------------------------------------------------------------------
__global__ __launch_bounds__(256) void pos_stage1(
    const float* __restrict__ viscen, const float* __restrict__ w,
    const float* __restrict__ bias, bf16* __restrict__ P1) {
  const int e = blockIdx.x * 256 + threadIdx.x;   // < MT*128
  const int r = e >> 7, c = e & 127;
  const float* vr = viscen + (size_t)r * 3;
  float x = vr[0] * w[c * 3 + 0] + vr[1] * w[c * 3 + 1] +
            vr[2] * w[c * 3 + 2] + bias[c];
  x = 0.5f * x * (1.f + erff(x * 0.70710678118f));
  P1[e] = (bf16)x;
}

__global__ __launch_bounds__(256) void add_pos_k(
    float* __restrict__ h, const float* __restrict__ pos) {
  const int e = blockIdx.x * 256 + threadIdx.x;
  h[e] += pos[e];
}

// ---------------------------------------------------------------------------
// LayerNorm over D_=384 (block per row, 128 threads x 3 elems).
// ---------------------------------------------------------------------------
template <bool OUT_BF16>
__global__ __launch_bounds__(128) void layernorm_k(
    const float* __restrict__ X, const float* __restrict__ g,
    const float* __restrict__ b, void* __restrict__ out) {
  const int r = blockIdx.x, t = threadIdx.x;
  const float* x = X + (size_t)r * D_;
  float v0 = x[t], v1 = x[t + 128], v2 = x[t + 256];
  __shared__ float rs[128], rss[128];
  rs[t] = v0 + v1 + v2;
  rss[t] = v0 * v0 + v1 * v1 + v2 * v2;
  __syncthreads();
  for (int o = 64; o > 0; o >>= 1) {
    if (t < o) { rs[t] += rs[t + o]; rss[t] += rss[t + o]; }
    __syncthreads();
  }
  __shared__ float mu_s, ri_s;
  if (t == 0) {
    float mu = rs[0] * (1.f / D_);
    float var = rss[0] * (1.f / D_) - mu * mu;
    mu_s = mu; ri_s = rsqrtf(var + EPS_);
  }
  __syncthreads();
  const float mu = mu_s, ri = ri_s;
#pragma unroll
  for (int i = 0; i < 3; ++i) {
    int c = t + i * 128;
    float y = (x[c] - mu) * ri * g[c] + b[c];
    if (OUT_BF16) ((bf16*)out)[(size_t)r * D_ + c] = (bf16)y;
    else          ((float*)out)[(size_t)r * D_ + c] = y;
  }
}

// ---------------------------------------------------------------------------
// Attention core: one block per (b, h). Nv=52, hd=64.
// ---------------------------------------------------------------------------
__global__ __launch_bounds__(128) void attention_k(
    const float* __restrict__ qkv, bf16* __restrict__ attn_out) {
  const int blk = blockIdx.x;            // b*6 + h
  const int b = blk / 6, h = blk % 6, t = threadIdx.x;
  __shared__ float Q[GV][64], Kt[GV][64], V[GV][64];
  __shared__ float S[GV][GV];
  const size_t base = (size_t)b * GV * 1152;
  for (int e = t; e < GV * 64; e += 128) {
    int i = e >> 6, d = e & 63;
    const float* row = qkv + base + (size_t)i * 1152 + h * 64 + d;
    Q[i][d]  = row[0];
    Kt[i][d] = row[384];
    V[i][d]  = row[768];
  }
  __syncthreads();
  for (int e = t; e < GV * GV; e += 128) {
    int i = e / GV, j = e % GV;
    float acc = 0.f;
#pragma unroll 8
    for (int d = 0; d < 64; ++d) acc += Q[i][d] * Kt[j][d];
    S[i][j] = acc * 0.125f;                       // hd^-0.5
  }
  __syncthreads();
  if (t < GV) {
    float mx = -3.4e38f;
    for (int j = 0; j < GV; ++j) mx = fmaxf(mx, S[t][j]);
    float sum = 0.f;
    for (int j = 0; j < GV; ++j) { float e2 = __expf(S[t][j] - mx); S[t][j] = e2; sum += e2; }
    float inv = 1.f / sum;
    for (int j = 0; j < GV; ++j) S[t][j] *= inv;
  }
  __syncthreads();
  for (int e = t; e < GV * 64; e += 128) {
    int i = e >> 6, d = e & 63;
    float acc = 0.f;
    for (int j = 0; j < GV; ++j) acc += S[i][j] * V[j][d];
    attn_out[((size_t)b * GV + i) * D_ + h * 64 + d] = (bf16)acc;
  }
}

// ---------------------------------------------------------------------------
// bf16 WMMA GEMM:  Out = epilogue(A[M,K] @ W[N,K]^T), W pre-converted bf16.
//   Block = 128 threads (4 waves), tile 128x64. Each wave: 32 rows x 64 cols
//   = 8x v_wmma_f32_16x16x32_bf16 per K-step. Double-buffered LDS: next
//   K-step's tiles are loaded to registers before the WMMA block and stashed
//   after it -> one barrier per K-step, global loads overlap the matrix pipe.
//   MODE: 0 f32 | 1 f32+bias | 2 bf16 bias+BN+ReLU | 3 bf16+bias
//         4 bf16 bias+GELU | 6 f32 bias+residual
//   CONCAT: A element (r,k) = k<256 ? FGb[r>>5][k] : A[r][k-256] (stage 3);
//           the 256 split is 8-element aligned so both sides vectorize.
// Requires M%128==0, N%64==0, K%32==0 (all shapes satisfy this).
// ---------------------------------------------------------------------------
#define AST 40
#define BST 40

template <int MODE, bool CONCAT>
__global__ __launch_bounds__(128) void gemm_wmma(
    const bf16* __restrict__ A, int lda,
    const bf16* __restrict__ W, int ldb,
    void* __restrict__ Out, int ldc,
    const float* __restrict__ bias,
    const float* __restrict__ bng, const float* __restrict__ bnb,
    const float* __restrict__ bnm, const float* __restrict__ bnv,
    const float* __restrict__ resid,
    const bf16* __restrict__ FGb,
    int M, int N, int Kd) {
  __shared__ bf16 As[2][128][AST];
  __shared__ bf16 Bs[2][64][BST];
  const int t = threadIdx.x;              // 128
  const int n0 = blockIdx.x * 64;
  const int m0 = blockIdx.y * 128;
  const int wave = t >> 5, lane = t & 31;
  const int lmod = lane & 15, khalf = lane >> 4;

  v8f acc[2][4];
#pragma unroll
  for (int a = 0; a < 2; ++a)
#pragma unroll
    for (int n = 0; n < 4; ++n)
#pragma unroll
      for (int e = 0; e < 8; ++e) acc[a][n][e] = 0.f;

  uint4 aR[4], bR[2];
  auto loadTiles = [&](int k0) {
#pragma unroll
    for (int i = 0; i < 4; ++i) {
      int c = t + (i << 7);                 // 0..511 chunks of 8 bf16
      int row = c >> 2, cc = (c & 3) << 3;
      const bf16* src;
      if (CONCAT) {
        int kk = k0 + cc;
        size_t grow = (size_t)(m0 + row);
        src = (kk < 256) ? FGb + (grow >> 5) * 256 + kk
                         : A + grow * (size_t)lda + (kk - 256);
      } else {
        src = A + (size_t)(m0 + row) * lda + k0 + cc;
      }
      aR[i] = *(const uint4*)src;
    }
#pragma unroll
    for (int i = 0; i < 2; ++i) {
      int c = t + (i << 7);                 // 0..255 chunks
      int n = c >> 2, ck = (c & 3) << 3;
      bR[i] = *(const uint4*)(W + (size_t)(n0 + n) * ldb + k0 + ck);
    }
  };
  auto stashTiles = [&](int buf) {
#pragma unroll
    for (int i = 0; i < 4; ++i) {
      int c = t + (i << 7);
      *(uint4*)&As[buf][c >> 2][(c & 3) << 3] = aR[i];
    }
#pragma unroll
    for (int i = 0; i < 2; ++i) {
      int c = t + (i << 7);
      *(uint4*)&Bs[buf][c >> 2][(c & 3) << 3] = bR[i];
    }
  };

  loadTiles(0);
  stashTiles(0);
  __syncthreads();
  int cur = 0;
  for (int k0 = 0; k0 < Kd; k0 += 32) {
    const bool more = (k0 + 32) < Kd;
    if (more) loadTiles(k0 + 32);

    // A fragments (16x32 bf16): lane m=lmod, K runs {kh*8..+8},{16+kh*8..+8}
    v16bf af0, af1;
    const bf16* ar0 = &As[cur][wave * 32 + lmod][0];
    const bf16* ar1 = &As[cur][wave * 32 + 16 + lmod][0];
#pragma unroll
    for (int i = 0; i < 8; ++i) {
      af0[i]     = ar0[khalf * 8 + i];
      af0[8 + i] = ar0[16 + khalf * 8 + i];
      af1[i]     = ar1[khalf * 8 + i];
      af1[8 + i] = ar1[16 + khalf * 8 + i];
    }
    // B fragments (32x16 bf16): lane n=lmod, K runs {kh*8..+8},{16+kh*8..+8}
#pragma unroll
    for (int nt = 0; nt < 4; ++nt) {
      v16bf bv;
      const bf16* br = &Bs[cur][nt * 16 + lmod][0];
#pragma unroll
      for (int i = 0; i < 8; ++i) { bv[i] = br[khalf * 8 + i]; bv[8 + i] = br[16 + khalf * 8 + i]; }
      acc[0][nt] = __builtin_amdgcn_wmma_f32_16x16x32_bf16(
          false, af0, false, bv, (short)0, acc[0][nt], false, false);
      acc[1][nt] = __builtin_amdgcn_wmma_f32_16x16x32_bf16(
          false, af1, false, bv, (short)0, acc[1][nt], false, false);
    }

    if (more) stashTiles(cur ^ 1);
    __syncthreads();
    cur ^= 1;
  }

  // Epilogue. C/D layout: VGPR j -> M = j + 8*khalf, N = lmod (+16*nt)
#pragma unroll
  for (int mt = 0; mt < 2; ++mt) {
    const int rbase = m0 + wave * 32 + mt * 16 + (khalf ? 8 : 0);
#pragma unroll
    for (int nt = 0; nt < 4; ++nt) {
      const int col = n0 + nt * 16 + lmod;
      float bv = 0.f, bns = 1.f, bno = 0.f;
      if (MODE != 0) bv = bias[col];
      if (MODE == 2) {
        bns = rsqrtf(bnv[col] + EPS_) * bng[col];
        bno = bnb[col] - bnm[col] * bns;
      }
#pragma unroll
      for (int j = 0; j < 8; ++j) {
        const int row = rbase + j;
        float v = acc[mt][nt][j] + bv;
        if (MODE == 2) v = fmaxf(v * bns + bno, 0.f);
        if (MODE == 4) v = 0.5f * v * (1.f + erff(v * 0.70710678118f));
        if (MODE == 6) v += resid[(size_t)row * ldc + col];
        const size_t o = (size_t)row * ldc + col;
        if (MODE == 2 || MODE == 3 || MODE == 4) ((bf16*)Out)[o] = (bf16)v;
        else                                     ((float*)Out)[o] = v;
      }
    }
  }
}

// ---------------------------------------------------------------------------
// Host orchestration
// ---------------------------------------------------------------------------
static inline size_t alignup(size_t x) { return (x + 255) & ~(size_t)255; }

extern "C" void kernel_launch(void* const* d_in, const int* in_sizes, int n_in,
                              void* d_out, int out_size, void* d_ws, size_t ws_size,
                              hipStream_t stream) {
  (void)in_sizes; (void)n_in; (void)out_size; (void)ws_size;
  const float* xyz    = (const float*)d_in[0];
  const int*   vis    = (const int*)  d_in[1];
  const float* enc_w1 = (const float*)d_in[2];
  const float* enc_b1 = (const float*)d_in[3];
  const float* bn1_g  = (const float*)d_in[4];
  const float* bn1_b  = (const float*)d_in[5];
  const float* bn1_m  = (const float*)d_in[6];
  const float* bn1_v  = (const float*)d_in[7];
  const float* enc_w2 = (const float*)d_in[8];
  const float* enc_b2 = (const float*)d_in[9];
  const float* enc_w3 = (const float*)d_in[10];
  const float* enc_b3 = (const float*)d_in[11];
  const float* bn2_g  = (const float*)d_in[12];
  const float* bn2_b  = (const float*)d_in[13];
  const float* bn2_m  = (const float*)d_in[14];
  const float* bn2_v  = (const float*)d_in[15];
  const float* enc_w4 = (const float*)d_in[16];
  const float* enc_b4 = (const float*)d_in[17];
  const float* pos_w1 = (const float*)d_in[18];
  const float* pos_b1 = (const float*)d_in[19];
  const float* pos_w2 = (const float*)d_in[20];
  const float* pos_b2 = (const float*)d_in[21];
  const float* ln1_g  = (const float*)d_in[22];
  const float* ln1_b  = (const float*)d_in[23];
  const float* w_qkv  = (const float*)d_in[24];
  const float* w_proj = (const float*)d_in[25];
  const float* b_proj = (const float*)d_in[26];
  const float* ln2_g  = (const float*)d_in[27];
  const float* ln2_b  = (const float*)d_in[28];
  const float* w_fc1  = (const float*)d_in[29];
  const float* b_fc1  = (const float*)d_in[30];
  const float* w_fc2  = (const float*)d_in[31];
  const float* b_fc2  = (const float*)d_in[32];
  const float* norm_g = (const float*)d_in[33];
  const float* norm_b = (const float*)d_in[34];

  // workspace arena
  uint8_t* base = (uint8_t*)d_ws;
  size_t o = 0;
  auto take = [&](size_t bytes) { size_t r = o; o = alignup(o + bytes); return r; };
  int*   cidx   = (int*)  (base + take((size_t)B_ * G_ * 4));
  float* center = (float*)(base + take((size_t)B_ * G_ * 3 * 4));
  float* neigh  = (float*)(base + take((size_t)B_ * G_ * KNN_ * 3 * 4));
  bf16*  X1     = (bf16*) (base + take((size_t)CR * 128 * 2));
  bf16*  X2     = (bf16*) (base + take((size_t)CR * 256 * 2));
  bf16*  FGb    = (bf16*) (base + take((size_t)CGRP * 256 * 2));
  bf16*  X3     = (bf16*) (base + take((size_t)CR * 512 * 2));
  bf16*  X4     = (bf16*) (base + take((size_t)CR * 384 * 2));
  float* tokens = (float*)(base + take((size_t)B_ * G_ * D_ * 4));
  float* viscen = (float*)(base + take((size_t)MT * 3 * 4));
  float* h      = (float*)(base + take((size_t)MT * D_ * 4));
  float* pos    = (float*)(base + take((size_t)MT * D_ * 4));
  bf16*  ybf    = (bf16*) (base + take((size_t)MT * D_ * 2));
  float* qkvb   = (float*)(base + take((size_t)MT * 1152 * 4));
  bf16*  attnbf = (bf16*) (base + take((size_t)MT * D_ * 2));
  bf16*  mlpbf  = (bf16*) (base + take((size_t)MT * 1536 * 2));
  bf16*  P1     = (bf16*) (base + take((size_t)MT * 128 * 2));
  // bf16 weight arenas
  bf16* w2b   = (bf16*)(base + take((size_t)256 * 128 * 2));
  bf16* w3b   = (bf16*)(base + take((size_t)512 * 512 * 2));
  bf16* w4b   = (bf16*)(base + take((size_t)D_ * 512 * 2));
  bf16* pw2b  = (bf16*)(base + take((size_t)D_ * 128 * 2));
  bf16* wqb   = (bf16*)(base + take((size_t)DEPTH_ * 3 * D_ * D_ * 2));
  bf16* wpb   = (bf16*)(base + take((size_t)DEPTH_ * D_ * D_ * 2));
  bf16* wf1b  = (bf16*)(base + take((size_t)DEPTH_ * 4 * D_ * D_ * 2));
  bf16* wf2b  = (bf16*)(base + take((size_t)DEPTH_ * 4 * D_ * D_ * 2));

  // ---- one-shot weight conversion (element counts all % 1024 == 0) ----
  auto cvt = [&](const float* s, bf16* d, size_t n) {
    cvt_bf16_k<<<(unsigned)(n / 1024), 256, 0, stream>>>(s, d);
  };
  cvt(enc_w2, w2b, (size_t)256 * 128);
  cvt(enc_w3, w3b, (size_t)512 * 512);
  cvt(enc_w4, w4b, (size_t)D_ * 512);
  cvt(pos_w2, pw2b, (size_t)D_ * 128);
  cvt(w_qkv,  wqb,  (size_t)DEPTH_ * 3 * D_ * D_);
  cvt(w_proj, wpb,  (size_t)DEPTH_ * D_ * D_);
  cvt(w_fc1,  wf1b, (size_t)DEPTH_ * 4 * D_ * D_);
  cvt(w_fc2,  wf2b, (size_t)DEPTH_ * 4 * D_ * D_);

  // ---- grouping ----
  fps_kernel<<<B_, 256, 0, stream>>>(xyz, cidx, center);
  knn_kernel<<<B_ * G_, 128, 0, stream>>>(xyz, center, neigh);

  // ---- PointNet encoder, 4 chunks of 8 batches ----
  for (int ch = 0; ch < 4; ++ch) {
    const int rowBase = ch * CR;
    enc_stage1<<<(CR * 128) / 256, 256, 0, stream>>>(
        neigh, enc_w1, enc_b1, bn1_g, bn1_b, bn1_m, bn1_v, X1, rowBase);
    gemm_wmma<3, false><<<dim3(256 / 64, CR / 128), 128, 0, stream>>>(
        X1, 128, w2b, 128, X2, 256, enc_b2,
        nullptr, nullptr, nullptr, nullptr, nullptr, nullptr, CR, 256, 128);
    maxpool_k<true><<<(CGRP * 256) / 256, 256, 0, stream>>>(X2, FGb, 256, 0);
    gemm_wmma<2, true><<<dim3(512 / 64, CR / 128), 128, 0, stream>>>(
        X2, 256, w3b, 512, X3, 512, enc_b3,
        bn2_g, bn2_b, bn2_m, bn2_v, nullptr, FGb, CR, 512, 512);
    gemm_wmma<3, false><<<dim3(384 / 64, CR / 128), 128, 0, stream>>>(
        X3, 512, w4b, 512, X4, 384, enc_b4,
        nullptr, nullptr, nullptr, nullptr, nullptr, nullptr, CR, 384, 512);
    maxpool_k<false><<<(CGRP * 384) / 256, 256, 0, stream>>>(X4, tokens, 384, ch * CGRP);
  }

  // ---- visible tokens + positional embedding ----
  gather_tokens<<<(MT * D_) / 256, 256, 0, stream>>>(tokens, center, vis, h, viscen);
  pos_stage1<<<(MT * 128) / 256, 256, 0, stream>>>(viscen, pos_w1, pos_b1, P1);
  gemm_wmma<1, false><<<dim3(384 / 64, MT / 128), 128, 0, stream>>>(
      P1, 128, pw2b, 128, pos, 384, pos_b2,
      nullptr, nullptr, nullptr, nullptr, nullptr, nullptr, MT, 384, 128);

  // ---- 12 transformer blocks ----
  for (int l = 0; l < DEPTH_; ++l) {
    const float* l1g = ln1_g + (size_t)l * D_;
    const float* l1b = ln1_b + (size_t)l * D_;
    const bf16*  wq  = wqb  + (size_t)l * 3 * D_ * D_;
    const bf16*  wp  = wpb  + (size_t)l * D_ * D_;
    const float* bp  = b_proj + (size_t)l * D_;
    const float* l2g = ln2_g + (size_t)l * D_;
    const float* l2b = ln2_b + (size_t)l * D_;
    const bf16*  wf1 = wf1b + (size_t)l * 4 * D_ * D_;
    const float* bf1 = b_fc1 + (size_t)l * 4 * D_;
    const bf16*  wf2 = wf2b + (size_t)l * 4 * D_ * D_;
    const float* bf2 = b_fc2 + (size_t)l * D_;

    add_pos_k<<<(MT * D_) / 256, 256, 0, stream>>>(h, pos);
    layernorm_k<true><<<MT, 128, 0, stream>>>(h, l1g, l1b, ybf);
    gemm_wmma<0, false><<<dim3(1152 / 64, MT / 128), 128, 0, stream>>>(
        ybf, 384, wq, 384, qkvb, 1152, nullptr,
        nullptr, nullptr, nullptr, nullptr, nullptr, nullptr, MT, 1152, 384);
    attention_k<<<B_ * 6, 128, 0, stream>>>(qkvb, attnbf);
    gemm_wmma<6, false><<<dim3(384 / 64, MT / 128), 128, 0, stream>>>(
        attnbf, 384, wp, 384, h, 384, bp,
        nullptr, nullptr, nullptr, nullptr, h, nullptr, MT, 384, 384);
    layernorm_k<true><<<MT, 128, 0, stream>>>(h, l2g, l2b, ybf);
    gemm_wmma<4, false><<<dim3(1536 / 64, MT / 128), 128, 0, stream>>>(
        ybf, 384, wf1, 384, mlpbf, 1536, bf1,
        nullptr, nullptr, nullptr, nullptr, nullptr, nullptr, MT, 1536, 384);
    gemm_wmma<6, false><<<dim3(384 / 64, MT / 128), 128, 0, stream>>>(
        mlpbf, 1536, wf2, 1536, h, 384, bf2,
        nullptr, nullptr, nullptr, nullptr, h, nullptr, MT, 384, 1536);
  }

  // ---- final layernorm -> d_out (f32) ----
  layernorm_k<false><<<MT, 128, 0, stream>>>(h, norm_g, norm_b, d_out);
}
